// CausalSelfAttention_25434796327765
// MI455X (gfx1250) — matrix operand order repaired and checked by reading, not compiled, not run on hardware
//
#include <hip/hip_runtime.h>
#include <hip/hip_bf16.h>

typedef __attribute__((ext_vector_type(16))) __bf16 v16bf;
typedef __attribute__((ext_vector_type(8)))  float  v8f;

#define D_MODEL 1024
#define NHEAD   16
#define DK      64
#define WINDOW  512
#define BSZ     2
#define TLEN    2048
#define MROWS   (BSZ * TLEN)   // 4096

// ---------------------------------------------------------------------------
// WMMA fragment helpers (wave32, v_wmma_f32_16x16x32_bf16)
// A layout (16x32, M x K): lane row = lane&15; k-chunks at 8*hi and 8*hi+16
// B layout (32x16, K x N): lane col = lane&15; k-chunks at 8*hi and 8*hi+16
// C/D layout (16x16):      lane n = lane&15; rows m = r + 8*hi in vgpr r
// ---------------------------------------------------------------------------
static __device__ inline v16bf load_fragA(const __bf16* base, int ldm, int lane) {
  const int row = lane & 15;
  const int kb  = (lane >> 4) << 3;
  const __bf16* p = base + (size_t)row * ldm + kb;
  union { v16bf v; uint4 u[2]; } f;
  f.u[0] = *(const uint4*)(p);
  f.u[1] = *(const uint4*)(p + 16);
  return f.v;
}

static __device__ inline v16bf load_fragB(const __bf16* base, int ldm, int lane) {
  const int col = lane & 15;
  const int kb  = (lane >> 4) << 3;
  const __bf16* p = base + (size_t)col * ldm + kb;
  union { v16bf v; uint4 u[2]; } f;
  f.u[0] = *(const uint4*)(p);
  f.u[1] = *(const uint4*)(p + 16);
  return f.v;
}

static __device__ inline v8f wmma_bf16(v16bf a, v16bf b, v8f c) {
  return __builtin_amdgcn_wmma_f32_16x16x32_bf16(false, a, false, b, (short)0, c,
                                                 false, false);
}

// ---------------------------------------------------------------------------
// Async DMA memory->LDS (CDNA5 Tensor/async path, tracked by ASYNCcnt).
// Low 32 bits of a generic pointer to __shared__ = wave-relative LDS offset
// (ISA 10.2 aperture mapping). offset:16 applies to BOTH lds and global sides
// (ISA 15.18.3 async pseudocode), so one addr pair moves 32 contiguous bytes.
// ---------------------------------------------------------------------------
static __device__ inline void async_tile32(uint32_t la, const __bf16* ga,
                                           uint32_t lb, const __bf16* gb) {
  asm volatile(
      "global_load_async_to_lds_b128 %0, %2, off\n\t"
      "global_load_async_to_lds_b128 %0, %2, off offset:16\n\t"
      "global_load_async_to_lds_b128 %1, %3, off\n\t"
      "global_load_async_to_lds_b128 %1, %3, off offset:16"
      :: "v"(la), "v"(lb),
         "v"((unsigned long long)(uintptr_t)ga),
         "v"((unsigned long long)(uintptr_t)gb)
      : "memory");
}

static __device__ inline void wait_async_le4() {
  asm volatile("s_wait_asynccnt 0x4" ::: "memory");
}
static __device__ inline void wait_async_0() {
  asm volatile("s_wait_asynccnt 0x0" ::: "memory");
}

// ---------------------------------------------------------------------------
// fp32 -> bf16 elementwise convert (x activations)
// ---------------------------------------------------------------------------
__global__ __launch_bounds__(256) void cvt_f32_bf16(const float* __restrict__ in,
                                                    __bf16* __restrict__ out, int n) {
  const int i = (blockIdx.x * 256 + threadIdx.x) * 4;
  if (i + 3 < n) {
    const float4 v = *(const float4*)(in + i);
    union { __bf16 h[4]; uint2 u; } o;
    o.h[0] = (__bf16)v.x; o.h[1] = (__bf16)v.y;
    o.h[2] = (__bf16)v.z; o.h[3] = (__bf16)v.w;
    *(uint2*)(out + i) = o.u;
  }
}

// ---------------------------------------------------------------------------
// fp32 W[k][n] -> bf16 WT[n][k]  (1024x1024), 32x32 LDS tiles
// ---------------------------------------------------------------------------
__global__ __launch_bounds__(256) void transpose_cvt(const float* __restrict__ W,
                                                     __bf16* __restrict__ WT) {
  __shared__ __bf16 t[32][33];
  const int bx = blockIdx.x * 32, by = blockIdx.y * 32;
  const int x = threadIdx.x, y = threadIdx.y;
  #pragma unroll
  for (int i = 0; i < 32; i += 8)
    t[y + i][x] = (__bf16)W[(size_t)(by + y + i) * D_MODEL + bx + x];
  __syncthreads();
  #pragma unroll
  for (int i = 0; i < 32; i += 8)
    WT[(size_t)(bx + y + i) * D_MODEL + by + x] = t[x][y + i];
}

// ---------------------------------------------------------------------------
// bf16 GEMM: out = A(4096x1024) * BT^T + bias ; BT is (N x K) row-major.
// Block tile 128x128, 8 waves of 32x64, K-step 32.
// Double-buffered LDS fed by global_load_async_to_lds_b128 (ASYNCcnt).
// mode 0: f32 row-major [4096,1024] (final projection)
// mode 1: bf16 [B,H,T,64]  (Q with scale=0.125, K with scale=1)
// mode 3: bf16 [B,H,64,T]  (V transposed for attention B-fragments)
// ---------------------------------------------------------------------------
__global__ __launch_bounds__(256) void gemm_bf16(const __bf16* __restrict__ A,
                                                 const __bf16* __restrict__ BT,
                                                 const float* __restrict__ bias,
                                                 void* __restrict__ out,
                                                 int mode, float scale) {
  constexpr int K = D_MODEL;
  __shared__ __bf16 sA[2][128 * 32];
  __shared__ __bf16 sB[2][128 * 32];
  const int tid   = threadIdx.x;
  const int lane  = tid & 31;
  const int wave  = tid >> 5;
  const int waveM = (wave >> 1) * 32;
  const int waveN = (wave & 1) * 64;
  const int bM = blockIdx.y * 128;
  const int bN = blockIdx.x * 128;

  v8f acc[2][4] = {};

  // each thread DMAs 32 contiguous bytes (16 bf16) of A and of B per K-tile
  const int lrow = tid >> 1;
  const int lcol = (tid & 1) * 16;
  const __bf16* gA = A  + (size_t)(bM + lrow) * K + lcol;
  const __bf16* gB = BT + (size_t)(bN + lrow) * K + lcol;
  uint32_t ldsA[2], ldsB[2];
  #pragma unroll
  for (int b = 0; b < 2; b++) {
    ldsA[b] = (uint32_t)(uintptr_t)&sA[b][lrow * 32 + lcol];
    ldsB[b] = (uint32_t)(uintptr_t)&sB[b][lrow * 32 + lcol];
  }

  async_tile32(ldsA[0], gA, ldsB[0], gB);  // preload tile 0

  int buf = 0;
  for (int k0 = 0; k0 < K; k0 += 32) {
    if (k0 + 32 < K) {
      async_tile32(ldsA[buf ^ 1], gA + k0 + 32, ldsB[buf ^ 1], gB + k0 + 32);
      wait_async_le4();   // current tile's 4 DMAs done; next tile's stay in flight
    } else {
      wait_async_0();
    }
    __syncthreads();      // tile 'buf' visible to all waves

    v16bf af[2], bfm[4];
    #pragma unroll
    for (int i = 0; i < 2; i++)
      af[i] = load_fragA(&sA[buf][(waveM + i * 16) * 32], 32, lane);
    #pragma unroll
    for (int j = 0; j < 4; j++)
      bfm[j] = load_fragB(&sB[buf][(waveN + j * 16) * 32], 32, lane);
    #pragma unroll
    for (int i = 0; i < 2; i++)
      #pragma unroll
      for (int j = 0; j < 4; j++)
        acc[i][j] = wmma_bf16(af[i], bfm[j], acc[i][j]);

    __syncthreads();      // all reads of 'buf' done before it is re-filled
    buf ^= 1;
  }

  const int ln = lane & 15;
  const int lm = (lane >> 4) * 8;
  #pragma unroll
  for (int j = 0; j < 4; j++) {
    const int   n  = bN + waveN + j * 16 + ln;
    const float bv = bias[n];
    #pragma unroll
    for (int i = 0; i < 2; i++) {
      const int m0 = bM + waveM + i * 16 + lm;
      if (mode == 3) {
        union { __bf16 h[8]; uint4 u; } pk;
        #pragma unroll
        for (int r = 0; r < 8; r++)
          pk.h[r] = (__bf16)((acc[i][j][r] + bv) * scale);
        const int b_ = m0 >> 11, t = m0 & 2047;
        const int h  = n >> 6,  d = n & 63;
        __bf16* o = (__bf16*)out;
        *(uint4*)&o[((size_t)((b_ * NHEAD + h) * DK + d)) * TLEN + t] = pk.u;
      } else if (mode == 0) {
        #pragma unroll
        for (int r = 0; r < 8; r++)
          ((float*)out)[(size_t)(m0 + r) * D_MODEL + n] =
              (acc[i][j][r] + bv) * scale;
      } else {
        #pragma unroll
        for (int r = 0; r < 8; r++) {
          const int m  = m0 + r;
          const int b_ = m >> 11, t = m & 2047;
          const int h  = n >> 6,  d = n & 63;
          ((__bf16*)out)[(((size_t)(b_ * NHEAD + h) * TLEN) + t) * DK + d] =
              (__bf16)((acc[i][j][r] + bv) * scale);
        }
      }
    }
  }
}

// ---------------------------------------------------------------------------
// Flash-style sliding-window attention. One wave = 16 queries, block = 8 waves
// = 128 queries; grid = B*H*(T/128) = 512 blocks. Q pre-scaled by 1/sqrt(dk).
// ---------------------------------------------------------------------------
__global__ __launch_bounds__(256) void attn_kernel(const __bf16* __restrict__ Q,
                                                   const __bf16* __restrict__ Km,
                                                   const __bf16* __restrict__ Vt,
                                                   __bf16* __restrict__ Oa) {
  __shared__ __bf16 sP[8][16 * 32];  // per-wave P bounce tile
  const int tid  = threadIdx.x;
  const int lane = tid & 31;
  const int wave = tid >> 5;
  const int bh = blockIdx.x >> 4;
  const int qc = blockIdx.x & 15;
  const int tq = qc * 128 + wave * 16;
  const __bf16* Qh = Q  + (size_t)bh * TLEN * DK;
  const __bf16* Kh = Km + (size_t)bh * TLEN * DK;
  const __bf16* Vh = Vt + (size_t)bh * DK * TLEN;

  const int ln = lane & 15;
  const int hi = lane >> 4;

  v16bf qa[2];
  #pragma unroll
  for (int dc = 0; dc < 2; dc++)
    qa[dc] = load_fragA(Qh + (size_t)tq * DK + dc * 32, DK, lane);

  float mrow[8], lrow[8];
  #pragma unroll
  for (int r = 0; r < 8; r++) { mrow[r] = -__builtin_inff(); lrow[r] = 0.0f; }
  v8f oacc[4] = {};

  int kb0 = tq - (WINDOW - 1);
  if (kb0 < 0) kb0 = 0;
  kb0 &= ~31;
  const int kbend = tq + 15;

  for (int kb = kb0; kb <= kbend; kb += 32) {
    // S = Q * K^T for 16 queries x 32 keys (two 16x16 C tiles)
    v8f S0 = {}, S1 = {};
    {
      const v16bf k00 = load_fragB(Kh + (size_t)kb * DK,             DK, lane);
      const v16bf k01 = load_fragB(Kh + (size_t)kb * DK + 32,        DK, lane);
      const v16bf k10 = load_fragB(Kh + (size_t)(kb + 16) * DK,      DK, lane);
      const v16bf k11 = load_fragB(Kh + (size_t)(kb + 16) * DK + 32, DK, lane);
      S0 = wmma_bf16(qa[0], k00, S0);
      S0 = wmma_bf16(qa[1], k01, S0);
      S1 = wmma_bf16(qa[0], k10, S1);
      S1 = wmma_bf16(qa[1], k11, S1);
    }

    float alpha_[8];
    #pragma unroll
    for (int r = 0; r < 8; r++) {
      const int  q  = tq + hi * 8 + r;
      const int  j0 = kb + ln;
      const int  j1 = kb + 16 + ln;
      const bool v0 = (j0 <= q) && (j0 >= q - (WINDOW - 1));
      const bool v1 = (j1 <= q) && (j1 >= q - (WINDOW - 1));
      float s0 = v0 ? S0[r] : -__builtin_inff();
      float s1 = v1 ? S1[r] : -__builtin_inff();
      float mx = fmaxf(s0, s1);
      #pragma unroll
      for (int off = 1; off < 16; off <<= 1)
        mx = fmaxf(mx, __shfl_xor(mx, off, 32));
      const float mnew = fmaxf(mrow[r], mx);
      const float p0 = v0 ? __expf(s0 - mnew) : 0.0f;
      const float p1 = v1 ? __expf(s1 - mnew) : 0.0f;
      float rs = p0 + p1;
      #pragma unroll
      for (int off = 1; off < 16; off <<= 1)
        rs += __shfl_xor(rs, off, 32);
      const float al = (mnew > -__builtin_inff()) ? __expf(mrow[r] - mnew) : 1.0f;
      lrow[r] = lrow[r] * al + rs;
      mrow[r] = mnew;
      alpha_[r] = al;
      // C-layout -> row-major 16x32 P tile in LDS (bf16)
      sP[wave][(hi * 8 + r) * 32 + ln]      = (__bf16)p0;
      sP[wave][(hi * 8 + r) * 32 + 16 + ln] = (__bf16)p1;
    }
    #pragma unroll
    for (int nb = 0; nb < 4; nb++)
      #pragma unroll
      for (int r = 0; r < 8; r++)
        oacc[nb][r] *= alpha_[r];

    // reload P as A-fragment (same-wave LDS ops stay in order)
    const v16bf pf = load_fragA(&sP[wave][0], 32, lane);
    #pragma unroll
    for (int nb = 0; nb < 4; nb++) {
      const v16bf vf = load_fragB(Vh + (size_t)(nb * 16) * TLEN + kb, TLEN, lane);
      oacc[nb] = wmma_bf16(pf, vf, oacc[nb]);
    }
  }

  const int b_ = bh >> 4;
  const int h  = bh & 15;
  float inv_[8];
  #pragma unroll
  for (int r = 0; r < 8; r++) inv_[r] = 1.0f / lrow[r];
  #pragma unroll
  for (int nb = 0; nb < 4; nb++)
    #pragma unroll
    for (int r = 0; r < 8; r++) {
      const int q = tq + hi * 8 + r;
      Oa[(size_t)(b_ * TLEN + q) * D_MODEL + h * DK + nb * 16 + ln] =
          (__bf16)(oacc[nb][r] * inv_[r]);
    }
}

// ---------------------------------------------------------------------------
extern "C" void kernel_launch(void* const* d_in, const int* in_sizes, int n_in,
                              void* d_out, int out_size, void* d_ws, size_t ws_size,
                              hipStream_t stream) {
  const float* x  = (const float*)d_in[0];
  const float* Wq = (const float*)d_in[1];
  const float* bq = (const float*)d_in[2];
  const float* Wk = (const float*)d_in[3];
  const float* bk = (const float*)d_in[4];
  const float* Wv = (const float*)d_in[5];
  const float* bv = (const float*)d_in[6];
  const float* Wo = (const float*)d_in[7];
  const float* bo = (const float*)d_in[8];

  char* ws = (char*)d_ws;
  size_t off = 0;
  __bf16* xb   = (__bf16*)(ws + off); off += (size_t)MROWS * D_MODEL * 2;    // 8 MB
  __bf16* wqt  = (__bf16*)(ws + off); off += (size_t)D_MODEL * D_MODEL * 2;  // 2 MB
  __bf16* wkt  = (__bf16*)(ws + off); off += (size_t)D_MODEL * D_MODEL * 2;
  __bf16* wvt  = (__bf16*)(ws + off); off += (size_t)D_MODEL * D_MODEL * 2;
  __bf16* wot  = (__bf16*)(ws + off); off += (size_t)D_MODEL * D_MODEL * 2;
  __bf16* qb   = (__bf16*)(ws + off); off += (size_t)MROWS * D_MODEL * 2;    // [B,H,T,64]
  __bf16* kbuf = (__bf16*)(ws + off); off += (size_t)MROWS * D_MODEL * 2;    // [B,H,T,64]
  __bf16* vbuf = (__bf16*)(ws + off); off += (size_t)MROWS * D_MODEL * 2;    // [B,H,64,T]
  __bf16* attb = (__bf16*)(ws + off); off += (size_t)MROWS * D_MODEL * 2;    // [B*T, D]

  cvt_f32_bf16<<<(MROWS * D_MODEL) / (256 * 4), 256, 0, stream>>>(
      x, xb, MROWS * D_MODEL);

  dim3 tb(32, 8), tg(32, 32);
  transpose_cvt<<<tg, tb, 0, stream>>>(Wq, wqt);
  transpose_cvt<<<tg, tb, 0, stream>>>(Wk, wkt);
  transpose_cvt<<<tg, tb, 0, stream>>>(Wv, wvt);
  transpose_cvt<<<tg, tb, 0, stream>>>(Wo, wot);

  dim3 gb(256), gg(D_MODEL / 128, MROWS / 128);  // (8, 32)
  gemm_bf16<<<gg, gb, 0, stream>>>(xb, wqt, bq, (void*)qb,   1, 0.125f);
  gemm_bf16<<<gg, gb, 0, stream>>>(xb, wkt, bk, (void*)kbuf, 1, 1.0f);
  gemm_bf16<<<gg, gb, 0, stream>>>(xb, wvt, bv, (void*)vbuf, 3, 1.0f);

  attn_kernel<<<BSZ * NHEAD * (TLEN / 128), 256, 0, stream>>>(qb, kbuf, vbuf, attb);

  gemm_bf16<<<gg, gb, 0, stream>>>(attb, wot, bo, d_out, 0, 1.0f);
}
